// VC_MOJI_Block_893353198473
// MI455X (gfx1250) — matrix operand order repaired and checked by reading, not compiled, not run on hardware
//
#include <hip/hip_runtime.h>

// ---------------------------------------------------------------------------
// MI455X (gfx1250) implementation. wave32, WMMA bf16 16x16x32.
// Pipeline: ECK gate -> 3x [LN -> QKV gemm -> window attn -> Wo gemm(+res)
//           -> LN -> W1 gemm(+gelu) -> W2 gemm(+res)] -> jitter -> VAE head
//           -> residual VQ -> decode broadcast add.
// GEMMs: bf16 WMMA, f32 accum, pre-transposed weights, double-buffered LDS
// staging via GLOBAL_LOAD_ASYNC_TO_LDS_B128 (ASYNCcnt) when available.
// ---------------------------------------------------------------------------

typedef __attribute__((ext_vector_type(16))) __bf16 v16bf;
typedef __attribute__((ext_vector_type(8)))  float  v8f;
typedef int v4i __attribute__((vector_size(16)));
typedef v4i __attribute__((address_space(1)))* v4i_g;   // global int4*
typedef v4i __attribute__((address_space(3)))* v4i_l;   // lds int4*

union FragU { v16bf v; unsigned u[8]; };

#define NTOK  16384     // B*T
#define DMODEL 768
#define TSEQ  2048
#define NBATCH 8

#if defined(__gfx1250__) && __has_builtin(__builtin_amdgcn_global_load_async_to_lds_b128) && __has_builtin(__builtin_amdgcn_s_wait_asynccnt)
#define USE_ASYNC_COPY 1
#else
#define USE_ASYNC_COPY 0
#endif

__device__ __forceinline__ void cp_async_b128(const void* g, void* l) {
#if USE_ASYNC_COPY
    __builtin_amdgcn_global_load_async_to_lds_b128(
        (v4i_g)(void*)g, (v4i_l)l, 0, 0);
#else
    *(uint4*)l = *(const uint4*)g;
#endif
}

template <int N>
__device__ __forceinline__ void wait_async() {
#if USE_ASYNC_COPY
    __builtin_amdgcn_s_wait_asynccnt(N);
#endif
}

__device__ __forceinline__ float gelu_f(float v) {
    const float c = 0.7978845608028654f;
    return 0.5f * v * (1.f + tanhf(c * (v + 0.044715f * v * v * v)));
}

__device__ __forceinline__ unsigned mixh(unsigned a, unsigned b, unsigned c) {
    unsigned h = a * 0x9E3779B1u ^ b * 0x85EBCA77u ^ c * 0xC2B2AE3Du;
    h ^= h >> 16; h *= 0x7FEB352Du; h ^= h >> 15; h *= 0x846CA68Bu; h ^= h >> 16;
    return h;
}
__device__ __forceinline__ float gaussf(unsigned a, unsigned b, unsigned c) {
    unsigned h1 = mixh(a, b, c), h2 = mixh(a ^ 0x68BC21EBu, b + 0x9E3779B9u, c);
    float u1 = (float)(h1 >> 8) * (1.f / 16777216.f) + 1e-7f;
    float u2 = (float)(h2 >> 8) * (1.f / 16777216.f);
    return sqrtf(-2.f * logf(u1)) * __cosf(6.28318530718f * u2);
}

// --------------------------------------------------------------------------
__global__ void init_kernel(float* acc, int n) {
    int i = blockIdx.x * blockDim.x + threadIdx.x;
    if (i < n) acc[i] = 0.f;
}

// f32 [K,N] -> bf16 transposed [N,K] (weights staged once per launch)
__global__ void cvt_bf16_t_kernel(const float* __restrict__ in, __bf16* __restrict__ out,
                                  int K, int N) {
    int i = blockIdx.x * blockDim.x + threadIdx.x;
    if (i >= K * N) return;
    int k = i / N, n = i % N;
    out[(size_t)n * K + k] = (__bf16)in[i];
}

// --------------------------------------------------------------------------
// ECKLock gate: gate = sigmoid( tanh(x@Wk) . key / sqrt(48) ); stream = x*gate
// --------------------------------------------------------------------------
__global__ __launch_bounds__(64) void eck_kernel(const float* __restrict__ x,
                                                 const float* __restrict__ Wk,
                                                 const float* __restrict__ key,
                                                 float* __restrict__ stream,
                                                 float* __restrict__ gsum) {
    __shared__ float accs[64];
    int row = blockIdx.x, tid = threadIdx.x;
    const float* xr = x + (size_t)row * DMODEL;
    float t = 0.f;
    if (tid < 48) {
        float s = 0.f;
        for (int i = 0; i < DMODEL; ++i) s += xr[i] * Wk[i * 48 + tid];
        t = tanhf(s) * key[tid];
    }
    accs[tid] = t;
    __syncthreads();
    for (int m = 32; m >= 1; m >>= 1) { if (tid < m) accs[tid] += accs[tid + m]; __syncthreads(); }
    float gate = 1.f / (1.f + __expf(-accs[0] * 0.14433756729740643f)); // 48^-0.5
    if (tid == 0) atomicAdd(gsum, gate);
    float* sr = stream + (size_t)row * DMODEL;
    for (int i = tid; i < DMODEL; i += 64) sr[i] = xr[i] * gate;
}

// --------------------------------------------------------------------------
// LayerNorm (fp32 stream -> bf16 operand). 8 waves / block, wave per token.
// --------------------------------------------------------------------------
__global__ __launch_bounds__(256) void ln_kernel(const float* __restrict__ x,
                                                 const float* __restrict__ g,
                                                 const float* __restrict__ b,
                                                 __bf16* __restrict__ out) {
    int wid = threadIdx.x >> 5, lane = threadIdx.x & 31;
    int tok = blockIdx.x * 8 + wid;
    const float* row = x + (size_t)tok * DMODEL;
    float s = 0.f, ss = 0.f;
    for (int i = lane; i < DMODEL; i += 32) { float v = row[i]; s += v; ss += v * v; }
    for (int m = 16; m >= 1; m >>= 1) { s += __shfl_xor(s, m, 32); ss += __shfl_xor(ss, m, 32); }
    float mean = s * (1.f / DMODEL);
    float var = ss * (1.f / DMODEL) - mean * mean;
    float inv = rsqrtf(var + 1e-5f);
    __bf16* o = out + (size_t)tok * DMODEL;
    for (int i = lane; i < DMODEL; i += 32)
        o[i] = (__bf16)((row[i] - mean) * inv * g[i] + b[i]);
}

// --------------------------------------------------------------------------
// bf16 WMMA GEMM: C[M,N] = epilogue(A[M,K] @ Bt[N,K]^T).
// Block tile 128x128x32, 8 waves (2x4), wave tile 64x32 (4x2 of 16x16).
// Both LDS tiles are [row][k] so K-pairs are contiguous for the CDNA5
// 16-bit A/B fragment layouts. Double-buffered, async-LDS staged.
// --------------------------------------------------------------------------
#define BM 128
#define BN 128
#define BK 32

__global__ __launch_bounds__(256) void gemm_bf16_kernel(
        const __bf16* __restrict__ A, const __bf16* __restrict__ Bt,
        int M, int N, int K,
        const float* __restrict__ bias, int use_gelu,
        float* __restrict__ resid, __bf16* __restrict__ outb) {
    __shared__ __align__(16) __bf16 As[2][BM * BK];
    __shared__ __align__(16) __bf16 Bs[2][BN * BK];

    int tid = threadIdx.x, lane = tid & 31, wid = tid >> 5;
    int n0 = blockIdx.x * BN, m0 = blockIdx.y * BM;
    int wm = (wid >> 2) * 64, wn = (wid & 3) * 32;
    int mloc = lane & 15, half = lane >> 4;

    v8f acc[4][2] = {};

    auto issue_tile = [&](int k0, int b) {
        #pragma unroll
        for (int i = 0; i < 2; ++i) {
            int lin = tid + i * 256;
            int r = lin >> 2, c8 = (lin & 3) * 8;
            cp_async_b128(&A[(size_t)(m0 + r) * K + k0 + c8], &As[b][r * BK + c8]);
            cp_async_b128(&Bt[(size_t)(n0 + r) * K + k0 + c8], &Bs[b][r * BK + c8]);
        }
    };

    const int ntile = K / BK;
    issue_tile(0, 0);
    int buf = 0;

    for (int t = 0; t < ntile; ++t) {
        if (t + 1 < ntile) {
            issue_tile((t + 1) * BK, buf ^ 1);
            wait_async<4>();          // newest tile (4 copies/thread) may remain in flight
        } else {
            wait_async<0>();
        }
        if (t + 2 < ntile) {          // light L2 prefetch two tiles ahead
            __builtin_prefetch(&A[(size_t)(m0 + (tid >> 1)) * K + (t + 2) * BK], 0, 1);
            __builtin_prefetch(&Bt[(size_t)(n0 + (tid >> 1)) * K + (t + 2) * BK], 0, 1);
        }
        __syncthreads();

        FragU af[4], bfr[2];
        #pragma unroll
        for (int mt = 0; mt < 4; ++mt) {
            int mrow = wm + mt * 16 + mloc;
            #pragma unroll
            for (int p = 0; p < 8; ++p) {
                int kk = half * 16 + 2 * p;
                af[mt].u[p] = *(const unsigned*)&As[buf][mrow * BK + kk];
            }
        }
        #pragma unroll
        for (int nt = 0; nt < 2; ++nt) {
            int ncol = wn + nt * 16 + mloc;
            #pragma unroll
            for (int p = 0; p < 8; ++p)
                bfr[nt].u[p] = *(const unsigned*)&Bs[buf][ncol * BK + half * 16 + 2 * p];
        }
        #pragma unroll
        for (int mt = 0; mt < 4; ++mt)
            #pragma unroll
            for (int nt = 0; nt < 2; ++nt)
                acc[mt][nt] = __builtin_amdgcn_wmma_f32_16x16x32_bf16(
                    false, af[mt].v, false, bfr[nt].v, (short)0, acc[mt][nt], false, false);
        __syncthreads();              // protect buf before it is refilled at t+2
        buf ^= 1;
    }

    #pragma unroll
    for (int mt = 0; mt < 4; ++mt)
        for (int nt = 0; nt < 2; ++nt)
            for (int r = 0; r < 8; ++r) {
                float v = acc[mt][nt][r];
                int gm = m0 + wm + mt * 16 + r + 8 * half;
                int gn = n0 + wn + nt * 16 + mloc;
                if (bias) v += bias[gn];
                if (use_gelu) v = gelu_f(v);
                size_t off = (size_t)gm * N + gn;
                if (resid) resid[off] += v;
                if (outb) outb[off] = (__bf16)v;
            }
}

// --------------------------------------------------------------------------
// Windowed attention. 2 waves / block, wave = one head of one (b, window).
// S = Q K^T (3x WMMA over dh=96), softmax across 16 lanes, O = A V
// (6x WMMA, K zero-padded 16->32). Head 0 reads K/V from the rolled window.
// --------------------------------------------------------------------------
__global__ __launch_bounds__(64) void attn_kernel(const __bf16* __restrict__ qkv,
                                                  __bf16* __restrict__ oattn) {
    __shared__ __align__(16) __bf16 q_s[2][16 * 96];
    __shared__ __align__(16) __bf16 k_s[2][16 * 96];
    __shared__ __align__(16) __bf16 vT[2][96 * 32];
    __shared__ __align__(16) __bf16 aS[2][16 * 32];

    int tid = threadIdx.x, lane = tid & 31, wid = tid >> 5;
    int blk = blockIdx.x;
    int hp = blk & 3;              // head pair
    int w  = (blk >> 2) & 127;     // window
    int b  = blk >> 9;             // batch
    int h  = hp * 2 + wid;
    int wk = (h == 0) ? ((w + 127) & 127) : w;   // n_ex = 1 -> head 0 rolled

    const size_t RS = 3 * DMODEL;  // qkv row stride (2304)
    size_t qb = (size_t)(b * TSEQ + w  * 16) * RS + h * 96;
    size_t kb = (size_t)(b * TSEQ + wk * 16) * RS + DMODEL + h * 96;
    size_t vb = (size_t)(b * TSEQ + wk * 16) * RS + 2 * DMODEL + h * 96;

    for (int i = lane; i < 16 * 96; i += 32) {
        int r = i / 96, c = i % 96;
        q_s[wid][i] = qkv[qb + (size_t)r * RS + c];
        k_s[wid][i] = qkv[kb + (size_t)r * RS + c];
    }
    for (int i = lane; i < 96 * 32; i += 32) vT[wid][i] = (__bf16)0.f;
    for (int i = lane; i < 16 * 32; i += 32) aS[wid][i] = (__bf16)0.f;
    __syncthreads();
    for (int i = lane; i < 16 * 96; i += 32) {
        int r = i / 96, c = i % 96;
        vT[wid][c * 32 + r] = qkv[vb + (size_t)r * RS + c];
    }
    __syncthreads();

    int mloc = lane & 15, half = lane >> 4;

    v8f s = {};
    #pragma unroll
    for (int ck = 0; ck < 3; ++ck) {
        int k0 = ck * 32;
        FragU a, kf;
        #pragma unroll
        for (int p = 0; p < 8; ++p) {
            int kk = half * 16 + 2 * p;
            a.u[p]  = *(const unsigned*)&q_s[wid][mloc * 96 + k0 + kk];
            kf.u[p] = *(const unsigned*)&k_s[wid][mloc * 96 + k0 + half * 16 + 2 * p];
        }
        s = __builtin_amdgcn_wmma_f32_16x16x32_bf16(false, a.v, false, kf.v, (short)0, s, false, false);
    }

    const float scale = 0.10206207261596575f;  // 96^-0.5
    #pragma unroll
    for (int r = 0; r < 8; ++r) {
        float v = s[r] * scale;
        float mx = v;
        for (int m = 8; m >= 1; m >>= 1) mx = fmaxf(mx, __shfl_xor(mx, m, 32));
        float e = __expf(v - mx);
        float su = e;
        for (int m = 8; m >= 1; m >>= 1) su += __shfl_xor(su, m, 32);
        aS[wid][(r + 8 * half) * 32 + mloc] = (__bf16)(e / su);
    }
    __syncthreads();

    FragU af;
    #pragma unroll
    for (int p = 0; p < 8; ++p) {
        int kk = half * 16 + 2 * p;
        af.u[p] = *(const unsigned*)&aS[wid][mloc * 32 + kk];
    }
    size_t ob = (size_t)(b * TSEQ + w * 16) * DMODEL + h * 96;
    #pragma unroll
    for (int nt = 0; nt < 6; ++nt) {
        FragU bv;
        #pragma unroll
        for (int p = 0; p < 8; ++p)
            bv.u[p] = *(const unsigned*)&vT[wid][(nt * 16 + mloc) * 32 + half * 16 + 2 * p];
        v8f o = {};
        o = __builtin_amdgcn_wmma_f32_16x16x32_bf16(false, af.v, false, bv.v, (short)0, o, false, false);
        #pragma unroll
        for (int r = 0; r < 8; ++r)
            oattn[ob + (size_t)(r + 8 * half) * DMODEL + nt * 16 + mloc] = (__bf16)o[r];
    }
}

// --------------------------------------------------------------------------
// ChannelJitterExchange (hash RNG stand-in for jax.random)
// --------------------------------------------------------------------------
__global__ __launch_bounds__(256) void cj_mean_kernel(float* __restrict__ chanSum) {
    __shared__ float red[256];
    int j = blockIdx.x >> 6, chunk = blockIdx.x & 63;
    int row = chunk * 256 + threadIdx.x;
    red[threadIdx.x] = 0.02f * gaussf((unsigned)row, (unsigned)j, 0x3C6EF372u);
    __syncthreads();
    for (int m = 128; m >= 1; m >>= 1) { if (threadIdx.x < m) red[threadIdx.x] += red[threadIdx.x + m]; __syncthreads(); }
    if (threadIdx.x == 0) atomicAdd(&chanSum[j], red[0]);
}

__global__ __launch_bounds__(128) void cj_apply_kernel(float* __restrict__ x,
                                                       const float* __restrict__ chanSum,
                                                       const float* __restrict__ cj_gate) {
    int row = blockIdx.x * 128 + threadIdx.x;
    if (row >= NTOK) return;
    float g = 1.f / (1.f + __expf(-cj_gate[0]));
    float* xr = x + (size_t)row * DMODEL;
    float vals[36];
    #pragma unroll
    for (int j = 0; j < 36; ++j) {
        int ch = (j * 97 + 13) % DMODEL;
        float nv = 0.02f * gaussf((unsigned)row, (unsigned)j, 0x3C6EF372u)
                 - chanSum[j] * (1.f / NTOK);
        float nx = xr[ch] + g * nv;
        xr[ch] = nx;
        vals[j] = nx;
    }
    #pragma unroll
    for (int j = 0; j < 36; ++j) {
        int ch = (j * 97 + 13) % DMODEL;
        xr[ch] = vals[(j * 5) % 36];
    }
}

// --------------------------------------------------------------------------
// VAE head
// --------------------------------------------------------------------------
__global__ void pool_kernel(const float* __restrict__ x, float* __restrict__ pooled) {
    int id = blockIdx.x * blockDim.x + threadIdx.x;
    if (id >= NBATCH * DMODEL) return;
    int b = id / DMODEL, d = id % DMODEL;
    float s = 0.f;
    const float* p = x + (size_t)b * TSEQ * DMODEL + d;
    for (int t = 0; t < TSEQ; ++t) s += p[(size_t)t * DMODEL];
    pooled[id] = s * (1.f / TSEQ);
}

__global__ void mulv_kernel(const float* __restrict__ pooled,
                            const float* __restrict__ Wmu, const float* __restrict__ bmu,
                            const float* __restrict__ Wlv, const float* __restrict__ blv,
                            float* __restrict__ mu, float* __restrict__ lv,
                            float* __restrict__ out_mu, float* __restrict__ out_lv) {
    int id = blockIdx.x * blockDim.x + threadIdx.x;
    if (id >= 1024) return;
    int kind = id >> 9, r = id & 511, b = r >> 6, j = r & 63;
    const float* W = kind ? Wlv : Wmu;
    const float* bb = kind ? blv : bmu;
    float s = bb[j];
    const float* p = pooled + b * DMODEL;
    for (int i = 0; i < DMODEL; ++i) s += p[i] * W[i * 64 + j];
    if (kind) { lv[r] = s; out_lv[r] = s; } else { mu[r] = s; out_mu[r] = s; }
}

__global__ void z_kernel(const float* __restrict__ mu, const float* __restrict__ lv,
                         float* __restrict__ z) {
    int id = blockIdx.x * blockDim.x + threadIdx.x;
    if (id >= 512) return;
    int b = id >> 6, j = id & 63;
    float eps = gaussf(0x7777u + b, 0x1234u + j, 0xABCDEFu);
    z[id] = mu[id] + __expf(0.5f * lv[id]) * eps;
}

// --------------------------------------------------------------------------
// Residual VQ (7 codebooks x 256 codes x 64 dims) + commit + contrastive
// --------------------------------------------------------------------------
__global__ __launch_bounds__(256) void quant_kernel(const float* __restrict__ z,
                                                    const float* __restrict__ codebooks,
                                                    float* __restrict__ quant_out,
                                                    float* __restrict__ commit_out,
                                                    float* __restrict__ contra_out) {
    __shared__ float zsh[512], res[512], qv[512];
    __shared__ float dist[256], red[256];
    __shared__ int bestI;
    __shared__ float nz[8], nq[8], logits[64], lse[8];
    int tid = threadIdx.x;
    for (int i = tid; i < 512; i += 256) { zsh[i] = z[i]; res[i] = z[i]; qv[i] = 0.f; }
    __syncthreads();
    for (int c = 0; c < 7; ++c) {
        for (int b = 0; b < 8; ++b) {
            const float* cb = codebooks + ((size_t)c * 256 + tid) * 64;
            const float* rb = &res[b * 64];
            float d = 0.f;
            for (int l = 0; l < 64; ++l) { float df = rb[l] - cb[l]; d += df * df; }
            dist[tid] = d;
            __syncthreads();
            if (tid == 0) {
                int bi = 0; float bd = dist[0];
                for (int i = 1; i < 256; ++i) if (dist[i] < bd) { bd = dist[i]; bi = i; }
                bestI = bi;
            }
            __syncthreads();
            if (tid < 64) {
                float cv = codebooks[((size_t)c * 256 + bestI) * 64 + tid];
                qv[b * 64 + tid] += cv;
                res[b * 64 + tid] -= cv;
            }
            __syncthreads();
        }
    }
    float p = 0.f;
    for (int i = tid; i < 512; i += 256) { float df = zsh[i] - qv[i]; p += df * df; }
    red[tid] = p; __syncthreads();
    for (int m = 128; m >= 1; m >>= 1) { if (tid < m) red[tid] += red[tid + m]; __syncthreads(); }
    if (tid == 0) *commit_out = red[0] * (1.f / 512.f);
    for (int i = tid; i < 512; i += 256) quant_out[i] = qv[i];  // quant_st value
    if (tid < 16) {
        int b = tid & 7;
        const float* v = (tid < 8) ? &zsh[b * 64] : &qv[b * 64];
        float s = 0.f;
        for (int l = 0; l < 64; ++l) s += v[l] * v[l];
        float nrm = sqrtf(s) + 1e-8f;
        if (tid < 8) nz[b] = nrm; else nq[b] = nrm;
    }
    __syncthreads();
    if (tid < 64) {
        int i = tid >> 3, j = tid & 7;
        float s = 0.f;
        for (int l = 0; l < 64; ++l) s += zsh[i * 64 + l] * qv[j * 64 + l];
        logits[tid] = s / (nz[i] * nq[j]) * 10.f;  // / TEMP
    }
    __syncthreads();
    if (tid < 8) {
        float mx = -1e30f;
        for (int j = 0; j < 8; ++j) mx = fmaxf(mx, logits[tid * 8 + j]);
        float s = 0.f;
        for (int j = 0; j < 8; ++j) s += __expf(logits[tid * 8 + j] - mx);
        lse[tid] = logf(s) + mx;
    }
    __syncthreads();
    if (tid == 0) {
        float s = 0.f;
        for (int i = 0; i < 8; ++i) s += lse[i] - logits[i * 8 + i];
        *contra_out = s * (1.f / 8.f);
    }
}

__global__ void delta_kernel(const float* __restrict__ quant, const float* __restrict__ We,
                             const float* __restrict__ be, float* __restrict__ delta) {
    int id = blockIdx.x * blockDim.x + threadIdx.x;
    if (id >= NBATCH * DMODEL) return;
    int b = id / DMODEL, d = id % DMODEL;
    float s = be[d];
    for (int j = 0; j < 64; ++j) s += quant[b * 64 + j] * We[j * DMODEL + d];
    delta[id] = s;
}

__global__ void bcast_add_kernel(float* __restrict__ x, const float* __restrict__ delta) {
    size_t i = (size_t)blockIdx.x * 256 + threadIdx.x;
    if (i >= (size_t)NTOK * DMODEL) return;
    int b = (int)(i / ((size_t)TSEQ * DMODEL));
    int d = (int)(i % DMODEL);
    x[i] += delta[b * DMODEL + d];
}

__global__ void finalize_kernel(const float* __restrict__ gsum,
                                const float* __restrict__ cj_gate,
                                float* __restrict__ scal) {
    scal[2] = 1.f / (1.f + __expf(-cj_gate[0]));     // g
    scal[3] = gsum[0] * (1.f / NTOK);                // eck_stats
}

// --------------------------------------------------------------------------
extern "C" void kernel_launch(void* const* d_in, const int* in_sizes, int n_in,
                              void* d_out, int out_size, void* d_ws, size_t ws_size,
                              hipStream_t stream) {
    const float* x_in   = (const float*)d_in[0];
    const float* eck_Wk = (const float*)d_in[1];
    const float* eck_ky = (const float*)d_in[2];
    const float* Wqkv   = (const float*)d_in[3];
    const float* Wo     = (const float*)d_in[4];
    const float* ln1_g  = (const float*)d_in[5];
    const float* ln1_b  = (const float*)d_in[6];
    const float* ln2_g  = (const float*)d_in[7];
    const float* ln2_b  = (const float*)d_in[8];
    const float* W1     = (const float*)d_in[9];
    const float* b1     = (const float*)d_in[10];
    const float* W2     = (const float*)d_in[11];
    const float* b2     = (const float*)d_in[12];
    const float* Wmu    = (const float*)d_in[13];
    const float* bmu    = (const float*)d_in[14];
    const float* Wlv    = (const float*)d_in[15];
    const float* blv    = (const float*)d_in[16];
    const float* cbooks = (const float*)d_in[17];
    const float* We     = (const float*)d_in[18];
    const float* be     = (const float*)d_in[19];
    const float* cjgate = (const float*)d_in[20];

    float* out = (float*)d_out;
    char*  ws  = (char*)d_ws;

    // d_out layout: x[12582912] mu[512] lv[512] commit contra g eck
    float* out_mu = out + (size_t)NTOK * DMODEL;
    float* out_lv = out_mu + 512;
    float* out_sc = out_lv + 512;   // [commit, contra, g, eck]

    // workspace layout (bytes); all weight mats stored transposed [N,K] bf16
    size_t o_xb   = 0;                                        // bf16 [NTOK,768]
    size_t o_wq   = o_xb   + (size_t)NTOK * DMODEL * 2;
    size_t o_wo   = o_wq   + (size_t)DMODEL * 2304 * 2;
    size_t o_w1   = o_wo   + (size_t)DMODEL * DMODEL * 2;
    size_t o_w2   = o_w1   + (size_t)DMODEL * 3072 * 2;
    size_t o_big  = o_w2   + (size_t)3072 * DMODEL * 2;       // union: qkv / mlp-mid
    size_t o_oatt = o_big  + (size_t)NTOK * 3072 * 2;
    size_t o_f32  = o_oatt + (size_t)NTOK * DMODEL * 2;

    __bf16* xb     = (__bf16*)(ws + o_xb);
    __bf16* wq_t   = (__bf16*)(ws + o_wq);   // [2304,768]
    __bf16* wo_t   = (__bf16*)(ws + o_wo);   // [768,768]
    __bf16* w1_t   = (__bf16*)(ws + o_w1);   // [3072,768]
    __bf16* w2_t   = (__bf16*)(ws + o_w2);   // [768,3072]
    __bf16* qkvb   = (__bf16*)(ws + o_big);  // [NTOK, 2304]
    __bf16* midb   = (__bf16*)(ws + o_big);  // [NTOK, 3072] (after attention done)
    __bf16* oattnb = (__bf16*)(ws + o_oatt); // [NTOK, 768]

    float* gsum    = (float*)(ws + o_f32);
    float* chanSum = gsum + 1;       // 36
    float* pooled  = chanSum + 36;   // 6144
    float* muv     = pooled + 6144;  // 512
    float* lvv     = muv + 512;      // 512
    float* zv      = lvv + 512;      // 512
    float* quantv  = zv + 512;       // 512
    float* deltav  = quantv + 512;   // 6144

    init_kernel<<<1, 64, 0, stream>>>(gsum, 37);

    // weight conversion + transpose to bf16 [N,K]
    {
        int n;
        n = DMODEL * 2304; cvt_bf16_t_kernel<<<(n + 255) / 256, 256, 0, stream>>>(Wqkv, wq_t, DMODEL, 2304);
        n = DMODEL * DMODEL; cvt_bf16_t_kernel<<<(n + 255) / 256, 256, 0, stream>>>(Wo, wo_t, DMODEL, DMODEL);
        n = DMODEL * 3072; cvt_bf16_t_kernel<<<(n + 255) / 256, 256, 0, stream>>>(W1, w1_t, DMODEL, 3072);
        n = 3072 * DMODEL; cvt_bf16_t_kernel<<<(n + 255) / 256, 256, 0, stream>>>(W2, w2_t, 3072, DMODEL);
    }

    eck_kernel<<<NTOK, 64, 0, stream>>>(x_in, eck_Wk, eck_ky, out, gsum);

    for (int it = 0; it < 3; ++it) {
        ln_kernel<<<NTOK / 8, 256, 0, stream>>>(out, ln1_g, ln1_b, xb);
        gemm_bf16_kernel<<<dim3(2304 / BN, NTOK / BM), 256, 0, stream>>>(
            xb, wq_t, NTOK, 2304, DMODEL, nullptr, 0, nullptr, qkvb);
        attn_kernel<<<NBATCH * 128 * 4, 64, 0, stream>>>(qkvb, oattnb);
        gemm_bf16_kernel<<<dim3(DMODEL / BN, NTOK / BM), 256, 0, stream>>>(
            oattnb, wo_t, NTOK, DMODEL, DMODEL, nullptr, 0, out, nullptr);
        ln_kernel<<<NTOK / 8, 256, 0, stream>>>(out, ln2_g, ln2_b, xb);
        gemm_bf16_kernel<<<dim3(3072 / BN, NTOK / BM), 256, 0, stream>>>(
            xb, w1_t, NTOK, 3072, DMODEL, b1, 1, nullptr, midb);
        gemm_bf16_kernel<<<dim3(DMODEL / BN, NTOK / BM), 256, 0, stream>>>(
            midb, w2_t, NTOK, DMODEL, 3072, b2, 0, out, nullptr);
    }

    cj_mean_kernel<<<36 * 64, 256, 0, stream>>>(chanSum);
    cj_apply_kernel<<<NTOK / 128, 128, 0, stream>>>(out, chanSum, cjgate);

    pool_kernel<<<(NBATCH * DMODEL + 255) / 256, 256, 0, stream>>>(out, pooled);
    mulv_kernel<<<4, 256, 0, stream>>>(pooled, Wmu, bmu, Wlv, blv, muv, lvv, out_mu, out_lv);
    z_kernel<<<2, 256, 0, stream>>>(muv, lvv, zv);
    quant_kernel<<<1, 256, 0, stream>>>(zv, cbooks, quantv, out_sc + 0, out_sc + 1);
    delta_kernel<<<(NBATCH * DMODEL + 255) / 256, 256, 0, stream>>>(quantv, We, be, deltav);
    bcast_add_kernel<<<(NTOK * DMODEL + 255) / 256, 256, 0, stream>>>(out, deltav);
    finalize_kernel<<<1, 1, 0, stream>>>(gsum, cjgate, out_sc);
}
